// SE3Transformer_20976620274309
// MI455X (gfx1250) — matrix operand order repaired and checked by reading, not compiled, not run on hardware
//
#include <hip/hip_runtime.h>
#include <hip/hip_bf16.h>

typedef _Float16 v4h  __attribute__((ext_vector_type(4)));
typedef _Float16 v8h  __attribute__((ext_vector_type(8)));
typedef _Float16 v16h __attribute__((ext_vector_type(16)));
typedef float    v8f  __attribute__((ext_vector_type(8)));

#define ND        64
#define ED        32
#define IN_DIM    160
#define NLAYERS   3
#define EPS_LN    1e-5f
#define NEG_SLOPE 0.01f

// LDS budget (bytes)
//   weights: w1t f16 [3][32][160] = 30720, w2t f16 [3][32][32] = 6144,
//            b1 384, b2 384, ln_g 1920, ln_b 1920        -> 41472
//   per wave: xraw f32[16][128]=8192, ebuf f32[16][32]=2048,
//             xnorm f16[16][160]=5120, hbuf f16[16][32]=1024 -> 16384
#define WEIGHT_BYTES   41472
#define PERWAVE_BYTES  16384
#define WAVES_PER_BLK  8
#define TILES_PER_WAVE 4
#define SMEM_BYTES     (WEIGHT_BYTES + WAVES_PER_BLK * PERWAVE_BYTES)

// intra-wave LDS producer->consumer fence (in-order DS pipe + compiler fence)
#define WAVE_LDS_SYNC() asm volatile("s_wait_dscnt 0x0" ::: "memory")

static __device__ __forceinline__ v16h ld_frag(const _Float16* p_lo, const _Float16* p_hi) {
    v8h lo = *(const v8h*)p_lo;
    v8h hi = *(const v8h*)p_hi;
    v16h r;
#pragma unroll
    for (int i = 0; i < 8; ++i) { r[i] = lo[i]; r[i + 8] = hi[i]; }
    return r;
}

__global__ void __launch_bounds__(256)
se3_edge_mlp_wmma(const float* __restrict__ node_feat,   // [N][64]
                  const float* __restrict__ edge_w,      // [E][32]
                  const float* __restrict__ ln_g,        // [3][160]
                  const float* __restrict__ ln_b,        // [3][160]
                  const float* __restrict__ w1,          // [3][160][32]
                  const float* __restrict__ b1,          // [3][32]
                  const float* __restrict__ w2,          // [3][32][32]
                  const float* __restrict__ b2,          // [3][32]
                  const int*   __restrict__ edge_i,      // [E]
                  const int*   __restrict__ edge_j,      // [E]
                  float*       __restrict__ out,         // [E][32]
                  int E) {
    extern __shared__ float4 smem4[];
    char* smem = (char*)smem4;

    _Float16* w1t = (_Float16*)smem;                    // [3][32][160]  (N-major, K contiguous)
    _Float16* w2t = w1t + NLAYERS * 32 * 160;           // [3][32][32]
    float* b1s = (float*)(w2t + NLAYERS * 32 * 32);     // [3][32]
    float* b2s = b1s + NLAYERS * 32;                    // [3][32]
    float* lng = b2s + NLAYERS * 32;                    // [3][160]
    float* lnb = lng + NLAYERS * 160;                   // [3][160]

    const int tid = threadIdx.x;

    // ---- one-time weight staging: transpose + convert to f16 in LDS ----
    for (int idx = tid; idx < NLAYERS * 160 * 32; idx += 256) {
        int l = idx / (160 * 32);
        int r = idx - l * 160 * 32;
        int k = r >> 5, n = r & 31;                     // w1[l][k][n]
        w1t[(l * 32 + n) * 160 + k] = (_Float16)w1[idx];
    }
    for (int idx = tid; idx < NLAYERS * 32 * 32; idx += 256) {
        int l = idx / (32 * 32);
        int r = idx - l * 32 * 32;
        int k = r >> 5, n = r & 31;                     // w2[l][k][n]
        w2t[(l * 32 + n) * 32 + k] = (_Float16)w2[idx];
    }
    for (int idx = tid; idx < NLAYERS * 32; idx += 256) { b1s[idx] = b1[idx]; b2s[idx] = b2[idx]; }
    for (int idx = tid; idx < NLAYERS * 160; idx += 256) { lng[idx] = ln_g[idx]; lnb[idx] = ln_b[idx]; }
    __syncthreads();

    // ---- per-wave private buffers ----
    const int wave = tid >> 5;
    const int lane = tid & 31;
    const int hs   = lane >> 4;       // half-wave select (0/1)
    const int lrow = lane & 15;       // edge row within 16-edge tile / matrix row-col index

    char* wb = smem + WEIGHT_BYTES + wave * PERWAVE_BYTES;
    float*    xraw  = (float*)wb;                        // [16][128] gathered node feats (f32)
    float*    ebuf  = xraw + 16 * 128;                   // [16][32]  running edge features (f32)
    _Float16* xnorm = (_Float16*)(ebuf + 16 * 32);       // [16][160] LN'd activations (f16, A layout src)
    _Float16* hbuf  = xnorm + 16 * 160;                  // [16][32]  hidden (f16)

    const int koffA = hs * 16;  // f16 A-fragment K offsets: lanes 0-15 -> {0,8}, 16-31 -> {16,24}
    const int cb    = hs * 80;  // this half-wave's column range within the 160-wide row

    const long tile0 = ((long)blockIdx.x * WAVES_PER_BLK + wave) * TILES_PER_WAVE;

    for (int t = 0; t < TILES_PER_WAVE; ++t) {
        const long e0 = (tile0 + t) * 16;
        if (e0 >= E) break;
        WAVE_LDS_SYNC();

        // ---- gather: lane (hs,lrow) loads 64 node feats for edge row lrow, endpoint hs ----
        {
            int eidx = (int)e0 + lrow;
            if (eidx >= E) eidx = E - 1;
            int node = (hs == 0) ? edge_i[eidx] : edge_j[eidx];
            const float* src = node_feat + (long)node * ND;
            float* dst = xraw + lrow * 128 + hs * 64;
#pragma unroll
            for (int c = 0; c < 64; c += 4)
                *(float4*)(dst + c) = *(const float4*)(src + c);
            // edge feature tile: row lrow, 16 cols per half-wave
            const float* esrc = edge_w + (long)eidx * ED + hs * 16;
            float* edst = ebuf + lrow * 32 + hs * 16;
#pragma unroll
            for (int c = 0; c < 16; c += 4)
                *(float4*)(edst + c) = *(const float4*)(esrc + c);
            // prefetch next tile's edge row + indices (global_prefetch_b8)
            long ne = e0 + 16 + lrow;
            if (t + 1 < TILES_PER_WAVE && ne < E) {
                __builtin_prefetch(edge_w + ne * ED, 0, 0);
                __builtin_prefetch((hs == 0 ? edge_i : edge_j) + ne, 0, 0);
            }
        }
        WAVE_LDS_SYNC();

        for (int l = 0; l < NLAYERS; ++l) {
            // ---- LayerNorm stats: each half-wave sums its 80 cols, combine via lane-xor-16 ----
            float s = 0.f, ss = 0.f;
#pragma unroll
            for (int q = 0; q < 20; ++q) {
                int col = cb + q * 4;
                const float* rp = (col < 128) ? (xraw + lrow * 128 + col)
                                              : (ebuf + lrow * 32 + (col - 128));
                float4 v = *(const float4*)rp;
                s  += v.x + v.y + v.z + v.w;
                ss += v.x * v.x + v.y * v.y + v.z * v.z + v.w * v.w;
            }
            s  += __shfl_xor(s, 16, 32);
            ss += __shfl_xor(ss, 16, 32);
            const float mu = s * (1.f / IN_DIM);
            const float rs = rsqrtf(ss * (1.f / IN_DIM) - mu * mu + EPS_LN);

            // ---- normalize + affine -> f16 A-source buffer (float4 in, 4xf16 out) ----
            {
                const float* gp = lng + l * 160;
                const float* bp = lnb + l * 160;
#pragma unroll
                for (int q = 0; q < 20; ++q) {
                    int col = cb + q * 4;
                    const float* rp = (col < 128) ? (xraw + lrow * 128 + col)
                                                  : (ebuf + lrow * 32 + (col - 128));
                    float4 rv = *(const float4*)rp;
                    float4 gv = *(const float4*)(gp + col);
                    float4 bv = *(const float4*)(bp + col);
                    v4h o;
                    o[0] = (_Float16)((rv.x - mu) * rs * gv.x + bv.x);
                    o[1] = (_Float16)((rv.y - mu) * rs * gv.y + bv.y);
                    o[2] = (_Float16)((rv.z - mu) * rs * gv.z + bv.z);
                    o[3] = (_Float16)((rv.w - mu) * rs * gv.w + bv.w);
                    *(v4h*)(xnorm + lrow * 160 + col) = o;
                }
            }
            WAVE_LDS_SYNC();

            // ---- GEMM1: x[16x160] @ W1[160x32] via 5 K-slices x 2 N-halves of WMMA ----
            v8f acc0 = {}, acc1 = {};
            const _Float16* arow = xnorm + lrow * 160;
            const _Float16* bn0  = w1t + (l * 32 + lrow) * 160 + hs * 16;        // col n = lrow
            const _Float16* bn1  = w1t + (l * 32 + 16 + lrow) * 160 + hs * 16;   // col n = 16+lrow
#pragma unroll
            for (int c = 0; c < 5; ++c) {
                const int kb = c * 32;
                v16h a  = ld_frag(arow + kb + koffA, arow + kb + koffA + 8);
                v16h f0 = ld_frag(bn0 + kb, bn0 + kb + 8);
                v16h f1 = ld_frag(bn1 + kb, bn1 + kb + 8);
                acc0 = __builtin_amdgcn_wmma_f32_16x16x32_f16(false, a, false, f0, (short)0, acc0, false, false);
                acc1 = __builtin_amdgcn_wmma_f32_16x16x32_f16(false, a, false, f1, (short)0, acc1, false, false);
            }

            // ---- bias + LeakyReLU, store hidden as f16 (C layout: lane=col, vgpr=row) ----
            {
                const float bb0 = b1s[l * 32 + lrow];
                const float bb1 = b1s[l * 32 + 16 + lrow];
#pragma unroll
                for (int r = 0; r < 8; ++r) {
                    int m = r + hs * 8;
                    float h0 = acc0[r] + bb0; h0 = (h0 > 0.f) ? h0 : h0 * NEG_SLOPE;
                    float h1 = acc1[r] + bb1; h1 = (h1 > 0.f) ? h1 : h1 * NEG_SLOPE;
                    hbuf[m * 32 + lrow]      = (_Float16)h0;
                    hbuf[m * 32 + 16 + lrow] = (_Float16)h1;
                }
            }
            WAVE_LDS_SYNC();

            // ---- GEMM2: h[16x32] @ W2[32x32] (single K-slice, 2 WMMAs) ----
            v8f d0 = {}, d1 = {};
            {
                const _Float16* hrow = hbuf + lrow * 32;
                v16h a2 = ld_frag(hrow + koffA, hrow + koffA + 8);
                const _Float16* c0p = w2t + (l * 32 + lrow) * 32 + hs * 16;
                const _Float16* c1p = w2t + (l * 32 + 16 + lrow) * 32 + hs * 16;
                v16h g0 = ld_frag(c0p, c0p + 8);
                v16h g1 = ld_frag(c1p, c1p + 8);
                d0 = __builtin_amdgcn_wmma_f32_16x16x32_f16(false, a2, false, g0, (short)0, d0, false, false);
                d1 = __builtin_amdgcn_wmma_f32_16x16x32_f16(false, a2, false, g1, (short)0, d1, false, false);
            }

            // ---- residual update of edge features in LDS ----
            {
                const float c0 = b2s[l * 32 + lrow];
                const float c1 = b2s[l * 32 + 16 + lrow];
#pragma unroll
                for (int r = 0; r < 8; ++r) {
                    int m = r + hs * 8;
                    ebuf[m * 32 + lrow]      += d0[r] + c0;
                    ebuf[m * 32 + 16 + lrow] += d1[r] + c1;
                }
            }
            WAVE_LDS_SYNC();
        }

        // ---- write out 16x32 tile (lane: row lrow, 16 cols per half-wave) ----
        {
            long eidx = e0 + lrow;
            if (eidx < E) {
                float* op = out + eidx * ED + hs * 16;
                const float* ep = ebuf + lrow * 32 + hs * 16;
#pragma unroll
                for (int c = 0; c < 16; c += 4)
                    *(float4*)(op + c) = *(const float4*)(ep + c);
            }
        }
    }
}

extern "C" void kernel_launch(void* const* d_in, const int* in_sizes, int n_in,
                              void* d_out, int out_size, void* d_ws, size_t ws_size,
                              hipStream_t stream) {
    const float* node_feat = (const float*)d_in[0];
    const float* edge_w    = (const float*)d_in[1];
    const float* ln_g      = (const float*)d_in[2];
    const float* ln_b      = (const float*)d_in[3];
    const float* w1        = (const float*)d_in[4];
    const float* b1        = (const float*)d_in[5];
    const float* w2        = (const float*)d_in[6];
    const float* b2        = (const float*)d_in[7];
    const int*   edge_i    = (const int*)d_in[8];
    const int*   edge_j    = (const int*)d_in[9];
    float*       out       = (float*)d_out;

    const int E = in_sizes[8];                 // edge count
    const long tiles = ((long)E + 15) / 16;
    const long tiles_per_blk = (long)WAVES_PER_BLK * TILES_PER_WAVE;
    const int blocks = (int)((tiles + tiles_per_blk - 1) / tiles_per_blk);

    (void)hipFuncSetAttribute((const void*)se3_edge_mlp_wmma,
                              hipFuncAttributeMaxDynamicSharedMemorySize,
                              (int)SMEM_BYTES);

    se3_edge_mlp_wmma<<<blocks, 256, SMEM_BYTES, stream>>>(
        node_feat, edge_w, ln_g, ln_b, w1, b1, w2, b2, edge_i, edge_j, out, E);
}
